// Memory_topk_29884382445749
// MI455X (gfx1250) — matrix operand order repaired and checked by reading, not compiled, not run on hardware
//
#include <hip/hip_runtime.h>
#include <hip/hip_bf16.h>
#include <stdint.h>

typedef __attribute__((ext_vector_type(16))) _Float16 v16h;
typedef __attribute__((ext_vector_type(8)))  float    v8f;

#define TOPK 32

// Problem constants (from reference setup_inputs)
#define NB     2       // batch
#define DE     128     // D_e
#define DO     256     // D_o
#define NT     16      // T
#define NH     64      // H
#define NW     64      // W
#define HREF   32
#define WREF   32
#define HWREF  1024    // HREF*WREF
#define THWREF 16384   // NT*HREF*WREF
#define NKEY   128     // TOPK * s * s
#define CSTRIDE 65536  // NT*NH*NW (channel stride in m_in/m_out)

// ---------------------------------------------------------------------------
// Kernel 1: per-column top-32 of qk_ref^T.  qk_ref is [B, THWREF, HWREF];
// one thread per column, rows streamed so global reads are fully coalesced.
// ---------------------------------------------------------------------------
__global__ __launch_bounds__(128)
void topk_kernel(const float* __restrict__ qk_ref,
                 int*   __restrict__ idx_ws,   // [B, HWREF, TOPK]
                 float* __restrict__ out_idx,  // [B, HWREF, 8] (as float)
                 float* __restrict__ out_val)  // [B, HWREF, 8]
{
    __shared__ float svals[TOPK * 128];
    __shared__ int   sidx [TOPK * 128];

    const int tid = threadIdx.x;                 // 0..127
    const int b   = blockIdx.x >> 3;             // 8 blocks per batch
    const int col = ((blockIdx.x & 7) << 7) + tid;   // 0..1023
    const float* src = qk_ref + (size_t)b * THWREF * HWREF + col;

    #pragma unroll
    for (int i = 0; i < TOPK; ++i) {
        svals[i * 128 + tid] = -3.4e38f;
        sidx [i * 128 + tid] = 0;
    }
    float curMin = -3.4e38f;
    int   minSlot = 0;

    for (int r = 0; r < THWREF; ++r) {
        float v = src[(size_t)r * HWREF];
        if (v > curMin) {
            svals[minSlot * 128 + tid] = v;
            sidx [minSlot * 128 + tid] = r;
            float m = svals[tid];
            int   ms = 0;
            #pragma unroll
            for (int i = 1; i < TOPK; ++i) {
                float x = svals[i * 128 + tid];
                if (x < m) { m = x; ms = i; }
            }
            curMin = m; minSlot = ms;
        }
    }

    int* wsd = idx_ws + ((size_t)b * HWREF + col) * TOPK;
    #pragma unroll
    for (int i = 0; i < TOPK; ++i) wsd[i] = sidx[i * 128 + tid];

    float* oi = out_idx + ((size_t)b * HWREF + col) * 8;
    float* ov = out_val + ((size_t)b * HWREF + col) * 8;
    for (int rk = 0; rk < 8; ++rk) {
        float best = -3.4e38f;
        int   bs = 0;
        #pragma unroll
        for (int i = 0; i < TOPK; ++i) {
            float x = svals[i * 128 + tid];
            if (x > best) { best = x; bs = i; }
        }
        ov[rk] = best;
        oi[rk] = (float)sidx[bs * 128 + tid];
        svals[bs * 128 + tid] = -3.4e38f;
    }
}

// ---------------------------------------------------------------------------
// Kernel 2: per query-block attention, v_wmma_f32_16x16x32_f16 for both
// GEMMs, K/V gathered with global_load_async_to_lds_b64 (ASYNCcnt path).
// One block (128 threads = 4 waves) per (b, h_ref, w_ref).
// LDS: nbuf f32 [64 ch][128 n] 32K + qlds f16 4K + plds f32 8K = 44 KB.
// ---------------------------------------------------------------------------
__global__ __launch_bounds__(128)
void attn_kernel(const float* __restrict__ m_in,
                 const float* __restrict__ m_out,
                 const float* __restrict__ q_in,
                 const int*   __restrict__ idx_ws,
                 float*       __restrict__ mem_out)
{
    __shared__ float    nbuf[64 * NKEY];    // stage buffer [c_local][n], f32
    __shared__ _Float16 qlds[128 * 16];     // [c][16 padded queries]
    __shared__ float    plds[16 * NKEY];    // logits/probs [q][n]

    const int tid  = threadIdx.x;
    const int lane = tid & 31;
    const int wid  = tid >> 5;              // wave id 0..3
    const int hi   = lane >> 4;             // half-wave
    const int lm   = lane & 15;

    const int blk = blockIdx.x;             // b*1024 + h*32 + w
    const int b   = blk >> 10;
    const int h   = (blk >> 5) & 31;
    const int w   = blk & 31;

    const int* idxp = idx_ws + ((size_t)b * HWREF + (h * WREF + w)) * TOPK;

    // gather task decode: 64 (block,si) pixel-pairs, 2 threads per pair
    const int p    = tid & 63;
    const int kb   = p >> 1, si = p & 1;
    const int bi   = idxp[kb];              // memory-block index in [0,16384)
    const int tt   = bi >> 10, hr = (bi >> 5) & 31, wr = bi & 31;
    const int n0   = kb * 4 + si * 2;       // first of two contiguous keys
    const int csub = tid >> 6;              // which 32-channel sub-slice
    const int pixoff = ((tt * NH + (hr * 2 + si)) * NW) + (wr * 2);

    const unsigned long long kbase =
        (unsigned long long)(m_in  + (size_t)b * DE * CSTRIDE);
    const unsigned long long vbase =
        (unsigned long long)(m_out + (size_t)b * DO * CSTRIDE);

    // ---- stage Q (f16, queries padded to 16 columns) ----
    {
        int c = tid;   // 0..127
        #pragma unroll
        for (int qi = 0; qi < 16; ++qi) {
            float v = 0.0f;
            if (qi < 4) {
                int i = qi >> 1, j = qi & 1;
                v = q_in[(((size_t)b * DE + c) * NH + (h * 2 + i)) * NW + (w * 2 + j)];
            }
            qlds[c * 16 + qi] = (_Float16)v;
        }
    }

    v8f acc0 = {}, acc1 = {};               // key tiles wid and wid+4
    const float scale = 0.08838834764831845f;   // 1/sqrt(128)

    // ---- QK^T: keys async-gathered in two 64-channel chunks ----
    for (int ch = 0; ch < 2; ++ch) {
        for (int ci = 0; ci < 32; ++ci) {
            int c_local = csub * 32 + ci;
            int c       = ch * 64 + c_local;
            unsigned lds_off = (unsigned)(uintptr_t)&nbuf[c_local * NKEY + n0];
            int goff = 4 * (c * CSTRIDE + pixoff);
            asm volatile("global_load_async_to_lds_b64 %0, %1, %2"
                         :: "v"(lds_off), "v"(goff), "s"(kbase) : "memory");
            __builtin_prefetch((const float*)vbase + (size_t)c * CSTRIDE + pixoff, 0, 0);
        }
        asm volatile("s_wait_asynccnt 0x0" ::: "memory");
        __syncthreads();

        for (int kk = 0; kk < 2; ++kk) {    // two K=32 chunks of channels
            #pragma unroll
            for (int ti = 0; ti < 2; ++ti) {
                const int kt = wid + 4 * ti;        // key tile
                const int n  = kt * 16 + lm;        // A row = key
                v16h a, bq;
                #pragma unroll
                for (int e = 0; e < 16; ++e) {      // A 16x32 f16 layout
                    int kl = kk * 32 + ((e >> 3) << 4) + hi * 8 + (e & 7);
                    a[e] = (_Float16)nbuf[kl * NKEY + n];
                }
                #pragma unroll
                for (int e = 0; e < 16; ++e) {      // B 32x16 f16 layout
                    int kg = ch * 64 + kk * 32 + ((e >> 3) << 4) + hi * 8 + (e & 7);
                    bq[e] = qlds[kg * 16 + lm];
                }
                if (ti == 0)
                    acc0 = __builtin_amdgcn_wmma_f32_16x16x32_f16(
                        false, a, false, bq, (short)0, acc0, false, false);
                else
                    acc1 = __builtin_amdgcn_wmma_f32_16x16x32_f16(
                        false, a, false, bq, (short)0, acc1, false, false);
            }
        }
        __syncthreads();                    // nbuf reuse / plds ordering
    }

    #pragma unroll
    for (int v = 0; v < 8; ++v) {           // D: row=key (v+hi*8), col=query(lm)
        plds[lm * NKEY + wid * 16       + v + hi * 8] = acc0[v] * scale;
        plds[lm * NKEY + (wid + 4) * 16 + v + hi * 8] = acc1[v] * scale;
    }
    __syncthreads();

    // ---- softmax over the 128 keys; wave w owns query w ----
    if (wid < 4) {
        int q = wid;
        float x[4];
        float vmax = -3.4e38f;
        #pragma unroll
        for (int t = 0; t < 4; ++t) {
            x[t] = plds[q * NKEY + lane + 32 * t];
            vmax = fmaxf(vmax, x[t]);
        }
        for (int off = 16; off > 0; off >>= 1)
            vmax = fmaxf(vmax, __shfl_xor(vmax, off, 32));
        float sum = 0.0f;
        #pragma unroll
        for (int t = 0; t < 4; ++t) { x[t] = __expf(x[t] - vmax); sum += x[t]; }
        for (int off = 16; off > 0; off >>= 1)
            sum += __shfl_xor(sum, off, 32);
        float inv = 1.0f / sum;
        #pragma unroll
        for (int t = 0; t < 4; ++t)
            plds[q * NKEY + lane + 32 * t] = x[t] * inv;
    }
    __syncthreads();

    // ---- P V : four 64-channel stages of D_o, async-gathered into nbuf ----
    for (int st = 0; st < 4; ++st) {
        for (int ci = 0; ci < 32; ++ci) {
            int c_local = csub * 32 + ci;
            int c       = st * 64 + c_local;
            unsigned lds_off = (unsigned)(uintptr_t)&nbuf[c_local * NKEY + n0];
            int goff = 4 * (c * CSTRIDE + pixoff);
            asm volatile("global_load_async_to_lds_b64 %0, %1, %2"
                         :: "v"(lds_off), "v"(goff), "s"(vbase) : "memory");
        }
        asm volatile("s_wait_asynccnt 0x0" ::: "memory");
        __syncthreads();

        // wave `wid` computes output tile st*4+wid (c_o = st*64 + wid*16 + m)
        v8f acc = {};
        for (int kc = 0; kc < 4; ++kc) {    // K-dim = 128 keys
            v16h a, bp;
            const int cc = wid * 16 + lm;   // A row m -> stage-local channel
            #pragma unroll
            for (int e = 0; e < 16; ++e) {  // A = V^T tile: M=c_o, K=key
                int k = kc * 32 + ((e >> 3) << 4) + hi * 8 + (e & 7);
                a[e] = (_Float16)nbuf[cc * NKEY + k];
            }
            #pragma unroll
            for (int e = 0; e < 16; ++e) {  // B = P: K=key, N=query
                int k = kc * 32 + ((e >> 3) << 4) + hi * 8 + (e & 7);
                bp[e] = (_Float16)plds[lm * NKEY + k];
            }
            acc = __builtin_amdgcn_wmma_f32_16x16x32_f16(
                false, a, false, bp, (short)0, acc, false, false);
        }
        if (lm < 4) {                       // only real queries
            int i = lm >> 1, j = lm & 1;
            #pragma unroll
            for (int v = 0; v < 8; ++v) {
                int co = st * 64 + wid * 16 + v + hi * 8;
                mem_out[(((size_t)b * DO + co) * NH + (h * 2 + i)) * NW
                        + (w * 2 + j)] = acc[v];
            }
        }
        __syncthreads();                    // before next stage overwrites nbuf
    }
}

// ---------------------------------------------------------------------------
extern "C" void kernel_launch(void* const* d_in, const int* in_sizes, int n_in,
                              void* d_out, int out_size, void* d_ws, size_t ws_size,
                              hipStream_t stream) {
    const float* m_in   = (const float*)d_in[0];
    const float* m_out  = (const float*)d_in[1];
    const float* q_in   = (const float*)d_in[2];
    const float* qk_ref = (const float*)d_in[3];

    float* out = (float*)d_out;
    float* out_mem = out;                                    // 2*256*64*64
    float* out_idx = out + (size_t)NB * DO * NH * NW;        // 2*1024*8
    float* out_val = out_idx + (size_t)NB * HWREF * 8;       // 2*1024*8

    int* idx_ws = (int*)d_ws;                                // 2*1024*32 ints

    topk_kernel<<<dim3(NB * 8), dim3(128), 0, stream>>>(qk_ref, idx_ws, out_idx, out_val);
    attn_kernel<<<dim3(NB * HWREF), dim3(128), 0, stream>>>(m_in, m_out, q_in, idx_ws, out_mem);
}
